// Topo_Label_Score_32444182954663
// MI455X (gfx1250) — matrix operand order repaired and checked by reading, not compiled, not run on hardware
//
#include <hip/hip_runtime.h>

// ---------------------------------------------------------------------------
// Topo_Label_Score for MI455X (gfx1250, wave32, WMMA)
// Dominant op: Agh = (g / D) @ hbn  --  12288x12288x128 GEMM, g streamed once,
// split into bf16 hi+lo and fed to v_wmma_f32_16x16x32_bf16 (2 WMMAs/tile).
// hbn is pre-packed into the WMMA B-fragment layout (L2-resident, 3 MB), so
// the GEMM inner loop is pure global_load_b128 + v_wmma with no LDS staging
// and no barriers. Row sums (D) and label boundary flags fused into the same
// g pass.
// ---------------------------------------------------------------------------

constexpr int NN      = 12288;
constexpr int IN_DIM  = 128;
constexpr int TOPK    = 512;
constexpr int BINS    = 10;
constexpr int NLAB    = 8;
constexpr float ALPHA = 0.6f;
constexpr float BN_EPS = 1e-5f;
constexpr int KSTEP   = 32;
constexpr int NTILES  = IN_DIM / 16;   // 8 column tiles of the 16x128 strip
constexpr int NKB     = NN / KSTEP;    // 384 k-blocks

typedef __attribute__((ext_vector_type(16))) __bf16 v16bf;
typedef __attribute__((ext_vector_type(8)))  float  v8f;

union BFU { unsigned short u; __bf16 b; };
static __device__ __forceinline__ unsigned short bf_bits(float x) { BFU c; c.b = (__bf16)x; return c.u; }

union BVU { uint4 q[2]; v16bf v; };

// ---------------- workspace layout (bytes) ----------------
constexpr size_t O_MEAN = 0;                         // 128 f32
constexpr size_t O_VAR  = 512;                       // 128 f32
constexpr size_t O_SC   = 1024;                      // scalars
constexpr size_t O_BPK  = 4096;                      // N*128 bf16 bits, B-fragment order
constexpr size_t O_AGH  = O_BPK  + (size_t)NN * IN_DIM * 2;   // N*128 f32
constexpr size_t O_D    = O_AGH  + (size_t)NN * IN_DIM * 4;   // N f32
constexpr size_t O_BND  = O_D    + (size_t)NN * 4;
constexpr size_t O_PL   = O_BND  + (size_t)NN * 4;
constexpr size_t O_Z3   = O_PL   + (size_t)NN * 4;
constexpr size_t O_HET  = O_Z3   + (size_t)NN * 4;
constexpr size_t O_KEYS = O_HET  + (size_t)NN * 4;
constexpr size_t O_TIDX = O_KEYS + (size_t)NN * 4;   // K int
constexpr size_t O_TKEY = O_TIDX + (size_t)TOPK * 4; // K u32

// ---------------------------------------------------------------------------
// 1) BatchNorm stats: one block per feature column
// ---------------------------------------------------------------------------
__global__ __launch_bounds__(256) void bn_stats_kernel(const float* __restrict__ h,
                                                       float* __restrict__ mean,
                                                       float* __restrict__ var) {
  __shared__ float ssum[256];
  __shared__ float ssq[256];
  const int c = blockIdx.x;
  const int t = threadIdx.x;
  float s = 0.f, q = 0.f;
  for (int i = t; i < NN; i += 256) {
    float x = h[(size_t)i * IN_DIM + c];
    s += x; q += x * x;
  }
  ssum[t] = s; ssq[t] = q; __syncthreads();
  for (int k = 128; k > 0; k >>= 1) {
    if (t < k) { ssum[t] += ssum[t + k]; ssq[t] += ssq[t + k]; }
    __syncthreads();
  }
  if (t == 0) {
    float m = ssum[0] / (float)NN;
    mean[c] = m;
    var[c]  = ssq[0] / (float)NN - m * m;
  }
}

// ---------------------------------------------------------------------------
// 2) BatchNorm apply: write f32 hbn (to output) + bf16 hbn directly into the
//    WMMA B-fragment layout:
//      element (row k, col c) -> bpack[(((k/32)*8 + c/16)*32 + lane)*16 + j]
//      lane = (((k%32)/8)%2 ? 16 : 0) + (c%16),  j = (k%32)%8 + ((k%32)/16)*8
// ---------------------------------------------------------------------------
__global__ __launch_bounds__(256) void bn_apply_kernel(const float* __restrict__ h,
                                                       const float* __restrict__ mean,
                                                       const float* __restrict__ var,
                                                       const float* __restrict__ gamma,
                                                       const float* __restrict__ beta,
                                                       float* __restrict__ hbn,
                                                       unsigned short* __restrict__ bpack) {
  size_t idx = (size_t)blockIdx.x * 256 + threadIdx.x;
  if (idx >= (size_t)NN * IN_DIM) return;
  int i = (int)(idx >> 7);          // row (k index of B)
  int c = (int)(idx & (IN_DIM - 1));
  float y = (h[idx] - mean[c]) * rsqrtf(var[c] + BN_EPS) * gamma[c] + beta[c];
  hbn[idx] = y;

  int kb   = i >> 5;
  int kr   = i & 31;
  int lane = (((kr >> 3) & 1) ? 16 : 0) + (c & 15);
  int j    = (kr & 7) + ((kr >> 4) << 3);
  int t    = c >> 4;
  size_t p = (((size_t)kb * NTILES + t) * 32 + lane) * 16 + j;
  bpack[p] = bf_bits(y);
}

// ---------------------------------------------------------------------------
// 3) Fused GEMM: Agh = (g/D) @ hbn, plus D and boundary flags.
//    1 wave per 16-row strip; bf16 hi/lo split of g; B fragments loaded
//    pre-packed from global (L2-resident). No barriers in the k-loop.
// ---------------------------------------------------------------------------
__global__ __launch_bounds__(256) void agh_wmma_kernel(const float* __restrict__ g,
                                                       const unsigned short* __restrict__ bpack,
                                                       const int* __restrict__ labels,
                                                       float* __restrict__ Agh,
                                                       float* __restrict__ Dout,
                                                       float* __restrict__ bout) {
  __shared__ int sLab[NN];                          // 48 KB (of 320 KB WGP LDS)

  const int tid  = threadIdx.x;
  const int lane = tid & 31;
  const int wave = tid >> 5;

  for (int i = tid; i < NN; i += 256) sLab[i] = labels[i];

  const int strip = blockIdx.x * 8 + wave;          // 96 blocks * 8 waves = 768 strips
  const int r0    = strip * 16;
  const int m     = lane & 15;
  const int row   = r0 + m;
  const int kbase = (lane < 16) ? 0 : 8;            // 16-bit A-matrix K base per lane half
  const int nlo   = lane & 15;                      // B/C column within tile

  v8f acc[NTILES];
#pragma unroll
  for (int t = 0; t < NTILES; ++t) acc[t] = v8f{0.f,0.f,0.f,0.f,0.f,0.f,0.f,0.f};

  float rowsum = 0.f;
  int lmax = -1, lmin = NLAB + 1;
  const float* grow = g + (size_t)row * NN;
  const uint4* bb = (const uint4*)bpack;            // 16B granules of packed B

  __syncthreads(); // labels staged

  for (int kb = 0; kb < NKB; ++kb) {
    const int k0 = kb * KSTEP;

    __builtin_prefetch(grow + k0 + 4 * KSTEP, 0, 1); // -> global_prefetch_b8

    // ---- A: 16 f32 of g per lane (two contiguous 8-float runs), split hi/lo
    float af[16];
    {
      const float4* gp0 = (const float4*)(grow + k0 + kbase);
      const float4* gp1 = (const float4*)(grow + k0 + kbase + 16);
      float4* afv = (float4*)af;
      afv[0] = gp0[0]; afv[1] = gp0[1];
      afv[2] = gp1[0]; afv[3] = gp1[1];
    }

    // ---- B: preload ALL 8 tiles' fragments (coalesced b128 pairs)
    BVU bv[NTILES];
#pragma unroll
    for (int t = 0; t < NTILES; ++t) {
      size_t q = ((size_t)(kb * NTILES + t) * 32 + lane) * 2;
      bv[t].q[0] = bb[q];
      bv[t].q[1] = bb[q + 1];
    }

    v16bf ahi, alo;
#pragma unroll
    for (int j = 0; j < 16; ++j) {
      float gv = af[j];
      __bf16 bh = (__bf16)gv;
      ahi[j] = bh;
      alo[j] = (__bf16)(gv - (float)bh);
      rowsum += gv;
      if (gv > 0.f) {
        int kk = k0 + kbase + j + ((j >= 8) ? 8 : 0);
        int L = sLab[kk];
        lmax = (L > lmax) ? L : lmax;
        lmin = (L < lmin) ? L : lmin;
      }
    }

    // ---- 16-WMMA burst (8 col tiles x hi/lo), no interleaved waits
#pragma unroll
    for (int t = 0; t < NTILES; ++t) {
      acc[t] = __builtin_amdgcn_wmma_f32_16x16x32_bf16(false, ahi, false, bv[t].v,
                                                       (short)0, acc[t], false, false);
      acc[t] = __builtin_amdgcn_wmma_f32_16x16x32_bf16(false, alo, false, bv[t].v,
                                                       (short)0, acc[t], false, false);
    }
  }

  // ---- finalize D / boundary (each row split across lanes L and L^16)
  float tot = rowsum + __shfl_xor(rowsum, 16, 32);
  int gmx = max(lmax, __shfl_xor(lmax, 16, 32));
  int gmn = min(lmin, __shfl_xor(lmin, 16, 32));
  if (lane < 16) {
    Dout[row] = tot;
    bout[row] = (gmx >= 0 && gmx != gmn) ? 1.f : 0.f;
  }

  // ---- scale by 1/D and store Agh (C layout: VGPR i -> M=i / M=i+8)
#pragma unroll
  for (int i = 0; i < 8; ++i) {
    int M = i + ((lane & 16) ? 8 : 0);
    float dM = __shfl(tot, M, 32);
    float rcp = 1.0f / dM;
#pragma unroll
    for (int t = 0; t < NTILES; ++t) {
      Agh[(size_t)(r0 + M) * IN_DIM + t * 16 + nlo] = acc[t][i] * rcp;
    }
  }
}

// ---------------------------------------------------------------------------
// 4) Per-row features: Z1 (L1), Z3 (dot W), histogram entropy. 1 block/row.
// ---------------------------------------------------------------------------
__global__ __launch_bounds__(128) void row_feats_kernel(const float* __restrict__ hbn,
                                                        const float* __restrict__ agh,
                                                        const float* __restrict__ D,
                                                        const float* __restrict__ W,
                                                        const float* __restrict__ bptr,
                                                        float* __restrict__ pl,
                                                        float* __restrict__ z3,
                                                        float* __restrict__ het) {
  __shared__ float sred[128];
  __shared__ int bins[BINS];
  const int i = blockIdx.x;
  const int c = threadIdx.x;
  float hv = hbn[(size_t)i * IN_DIM + c];
  float av = agh[(size_t)i * IN_DIM + c];

  sred[c] = hv; __syncthreads();
  for (int s = 64; s > 0; s >>= 1) { if (c < s) sred[c] = fminf(sred[c], sred[c + s]); __syncthreads(); }
  float rmin = sred[0]; __syncthreads();
  sred[c] = hv; __syncthreads();
  for (int s = 64; s > 0; s >>= 1) { if (c < s) sred[c] = fmaxf(sred[c], sred[c + s]); __syncthreads(); }
  float rmax = sred[0]; __syncthreads();

  float width = (rmax - rmin > 0.f) ? (rmax - rmin) : 1.f;
  int bidx = (int)floorf((hv - rmin) / width * (float)BINS);
  bidx = min(max(bidx, 0), BINS - 1);
  if (c < BINS) bins[c] = 0;
  __syncthreads();
  atomicAdd(&bins[bidx], 1);

  sred[c] = fabsf(hv - av); __syncthreads();
  for (int s = 64; s > 0; s >>= 1) { if (c < s) sred[c] += sred[c + s]; __syncthreads(); }
  float z1 = sred[0]; __syncthreads();
  sred[c] = av * W[c]; __syncthreads();
  for (int s = 64; s > 0; s >>= 1) { if (c < s) sred[c] += sred[c + s]; __syncthreads(); }
  float z3v = sred[0];

  if (c == 0) {
    float pv[BINS], psum = 0.f;
#pragma unroll
    for (int t = 0; t < BINS; ++t) { pv[t] = (float)bins[t] / 128.f + 1e-10f; psum += pv[t]; }
    float H = 0.f;
#pragma unroll
    for (int t = 0; t < BINS; ++t) { float p = pv[t] / psum; H -= p * logf(p); }
    het[i] = H;
    pl[i]  = z1 + 1.f / (1.f + expf(-D[i]));
    z3[i]  = z3v + bptr[0];
  }
}

// ---------------------------------------------------------------------------
// 5) Global reductions: softmax max/sumexp over Z3, het min/max. One block.
// ---------------------------------------------------------------------------
__global__ __launch_bounds__(1024) void globals_kernel(const float* __restrict__ z3,
                                                       const float* __restrict__ het,
                                                       float* __restrict__ sc) {
  __shared__ float s[1024];
  const int t = threadIdx.x;
  float m = -3.4e38f;
  for (int i = t; i < NN; i += 1024) m = fmaxf(m, z3[i]);
  s[t] = m; __syncthreads();
  for (int k = 512; k > 0; k >>= 1) { if (t < k) s[t] = fmaxf(s[t], s[t + k]); __syncthreads(); }
  float z3max = s[0]; __syncthreads();

  float se = 0.f;
  for (int i = t; i < NN; i += 1024) se += expf(z3[i] - z3max);
  s[t] = se; __syncthreads();
  for (int k = 512; k > 0; k >>= 1) { if (t < k) s[t] += s[t + k]; __syncthreads(); }
  float sumexp = s[0]; __syncthreads();

  float hn = 3.4e38f;
  for (int i = t; i < NN; i += 1024) hn = fminf(hn, het[i]);
  s[t] = hn; __syncthreads();
  for (int k = 512; k > 0; k >>= 1) { if (t < k) s[t] = fminf(s[t], s[t + k]); __syncthreads(); }
  float hetmin = s[0]; __syncthreads();

  float hx = -3.4e38f;
  for (int i = t; i < NN; i += 1024) hx = fmaxf(hx, het[i]);
  s[t] = hx; __syncthreads();
  for (int k = 512; k > 0; k >>= 1) { if (t < k) s[t] = fmaxf(s[t], s[t + k]); __syncthreads(); }
  float hetmax = s[0];

  if (t == 0) { sc[0] = z3max; sc[1] = sumexp; sc[2] = hetmin; sc[3] = hetmax; }
}

// ---------------------------------------------------------------------------
// 6) Final scores -> monotonic uint keys (all scores positive)
// ---------------------------------------------------------------------------
__global__ __launch_bounds__(256) void score_kernel(const float* __restrict__ pl,
                                                    const float* __restrict__ z3,
                                                    const float* __restrict__ het,
                                                    const float* __restrict__ bnd,
                                                    const float* __restrict__ sc,
                                                    unsigned* __restrict__ keys) {
  int i = blockIdx.x * 256 + threadIdx.x;
  if (i >= NN) return;
  float pg = expf(z3[i] - sc[0]) / sc[1];
  float hetn = (het[i] - sc[2]) / (sc[3] - sc[2] + 1e-10f);
  float plab = ALPHA * bnd[i] + (1.f - ALPHA) * hetn;
  float pt = pl[i] + pg;
  float total = 1.f / (1.f + expf(-pt * plab));
  keys[i] = __float_as_uint(total);
}

// ---------------------------------------------------------------------------
// 7) Radix-select exact K-th largest key (1 block, 32 bit passes)
// ---------------------------------------------------------------------------
__global__ __launch_bounds__(256) void radix_kernel(const unsigned* __restrict__ keys,
                                                    unsigned* __restrict__ Tout) {
  __shared__ unsigned scnt[256];
  __shared__ unsigned sprefix;
  const int t = threadIdx.x;
  if (t == 0) sprefix = 0;
  __syncthreads();
  for (int bit = 31; bit >= 0; --bit) {
    unsigned cand = sprefix | (1u << bit);
    unsigned c = 0;
    for (int i = t; i < NN; i += 256) c += (keys[i] >= cand) ? 1u : 0u;
    scnt[t] = c; __syncthreads();
    for (int k = 128; k > 0; k >>= 1) { if (t < k) scnt[t] += scnt[t + k]; __syncthreads(); }
    if (t == 0 && scnt[0] >= (unsigned)TOPK) sprefix = cand;
    __syncthreads();
  }
  if (t == 0) Tout[0] = sprefix;
}

// ---------------------------------------------------------------------------
// 8) Gather the top-K (strictly greater, then ties)
// ---------------------------------------------------------------------------
__global__ __launch_bounds__(256) void gather_kernel(const unsigned* __restrict__ keys,
                                                     const unsigned* __restrict__ Tin,
                                                     int* __restrict__ topIdx,
                                                     unsigned* __restrict__ topKey) {
  __shared__ unsigned cg, ct, ngreater;
  const int t = threadIdx.x;
  unsigned T = Tin[0];
  if (t == 0) { cg = 0; ct = 0; }
  __syncthreads();
  for (int i = t; i < NN; i += 256)
    if (keys[i] > T) { unsigned p = atomicAdd(&cg, 1u); topIdx[p] = i; topKey[p] = keys[i]; }
  __syncthreads();
  if (t == 0) ngreater = cg;
  __syncthreads();
  for (int i = t; i < NN; i += 256)
    if (keys[i] == T) {
      unsigned p = atomicAdd(&ct, 1u);
      unsigned s = ngreater + p;
      if (s < (unsigned)TOPK) { topIdx[s] = i; topKey[s] = keys[i]; }
    }
}

// ---------------------------------------------------------------------------
// 9) Bitonic sort of 512 (key desc, index asc) -> write int indices to out
// ---------------------------------------------------------------------------
__global__ __launch_bounds__(512) void sort_kernel(const int* __restrict__ topIdx,
                                                   const unsigned* __restrict__ topKey,
                                                   int* __restrict__ outIdx) {
  __shared__ unsigned long long s[TOPK];
  const int t = threadIdx.x;
  s[t] = ((unsigned long long)topKey[t] << 32) |
         (unsigned long long)(0xFFFFFFFFu - (unsigned)topIdx[t]);
  __syncthreads();
  for (int k = 2; k <= TOPK; k <<= 1) {
    for (int j = k >> 1; j > 0; j >>= 1) {
      int ixj = t ^ j;
      if (ixj > t) {
        bool descRegion = ((t & k) == 0);
        unsigned long long a = s[t], b = s[ixj];
        if ((a < b) == descRegion) { s[t] = b; s[ixj] = a; }
      }
      __syncthreads();
    }
  }
  outIdx[t] = (int)(0xFFFFFFFFu - (unsigned)(s[t] & 0xFFFFFFFFull));
}

// ---------------------------------------------------------------------------
extern "C" void kernel_launch(void* const* d_in, const int* in_sizes, int n_in,
                              void* d_out, int out_size, void* d_ws, size_t ws_size,
                              hipStream_t stream) {
  const float* g      = (const float*)d_in[0];
  const float* h      = (const float*)d_in[1];
  const int*   labels = (const int*)d_in[2];
  const float* gamma  = (const float*)d_in[3];
  const float* beta   = (const float*)d_in[4];
  const float* W      = (const float*)d_in[5];
  const float* b      = (const float*)d_in[6];

  char* ws = (char*)d_ws;
  float*          wMean = (float*)(ws + O_MEAN);
  float*          wVar  = (float*)(ws + O_VAR);
  float*          wSc   = (float*)(ws + O_SC);
  unsigned short* wBpk  = (unsigned short*)(ws + O_BPK);
  float*          wAgh  = (float*)(ws + O_AGH);
  float*          wD    = (float*)(ws + O_D);
  float*          wBnd  = (float*)(ws + O_BND);
  float*          wPl   = (float*)(ws + O_PL);
  float*          wZ3   = (float*)(ws + O_Z3);
  float*          wHet  = (float*)(ws + O_HET);
  unsigned*       wKeys = (unsigned*)(ws + O_KEYS);
  int*            wTIdx = (int*)(ws + O_TIDX);
  unsigned*       wTKey = (unsigned*)(ws + O_TKEY);

  float* outG   = (float*)d_out;                         // N*N
  float* outHbn = outG + (size_t)NN * NN;                // N*128
  int*   outTop = (int*)(outHbn + (size_t)NN * IN_DIM);  // K

  // g passes through unchanged
  hipMemcpyAsync(outG, g, (size_t)NN * NN * sizeof(float),
                 hipMemcpyDeviceToDevice, stream);

  bn_stats_kernel<<<IN_DIM, 256, 0, stream>>>(h, wMean, wVar);
  bn_apply_kernel<<<(NN * IN_DIM) / 256, 256, 0, stream>>>(h, wMean, wVar, gamma, beta,
                                                           outHbn, wBpk);
  agh_wmma_kernel<<<NN / 16 / 8, 256, 0, stream>>>(g, wBpk, labels, wAgh, wD, wBnd);
  row_feats_kernel<<<NN, 128, 0, stream>>>(outHbn, wAgh, wD, W, b, wPl, wZ3, wHet);
  globals_kernel<<<1, 1024, 0, stream>>>(wZ3, wHet, wSc);
  score_kernel<<<NN / 256, 256, 0, stream>>>(wPl, wZ3, wHet, wBnd, wSc, wKeys);
  radix_kernel<<<1, 256, 0, stream>>>(wKeys, (unsigned*)(wSc + 8));
  gather_kernel<<<1, 256, 0, stream>>>(wKeys, (unsigned*)(wSc + 8), wTIdx, wTKey);
  sort_kernel<<<1, TOPK, 0, stream>>>(wTIdx, wTKey, outTop);
}
